// HierarchicalLatentBackbone_64759516889548
// MI455X (gfx1250) — matrix operand order repaired and checked by reading, not compiled, not run on hardware
//
#include <hip/hip_runtime.h>
#include <math.h>

// ---------------------------------------------------------------------------
// HierarchicalLatentBackbone forward for MI455X (gfx1250, wave32, WMMA).
// Heavy GEMMs use V_WMMA_F32_16X16X32_F16 with pre-transposed f16 weights
// (L2-resident). Each wave computes a 16x64 C tile: one A fragment (contiguous
// float4 loads + packed f16 converts) feeds 4 WMMAs; B fragments are
// contiguous 32B v16h loads. Split-A concat is handled by splitting the
// K-loop at K1 (multiple of 32) so the hot loop has no selects.
// ---------------------------------------------------------------------------

typedef __attribute__((ext_vector_type(16))) _Float16 v16h;
typedef __attribute__((ext_vector_type(8)))  float    v8f;
typedef __attribute__((ext_vector_type(4)))  float    f32x4;

__device__ __forceinline__ float geluf(float x) {
  return 0.5f * x * (1.0f + erff(x * 0.70710678118654752f));
}
__device__ __forceinline__ float halfsigf(float x) {
  return 0.5f + 0.5f / (1.0f + expf(-x));
}

// Transpose + fp32->f16 convert weights: Wt[n*K + k] = (f16) W[k*N + n]
__global__ __launch_bounds__(256)
void k_wcvt(const float* __restrict__ W, _Float16* __restrict__ Wt,
            int K, int N)
{
  size_t i = (size_t)blockIdx.x * blockDim.x + threadIdx.x;
  if (i >= (size_t)K * N) return;
  int k = (int)(i % K);
  int n = (int)(i / K);
  Wt[i] = (_Float16)W[(size_t)k * N + n];
}

// ---------------------------------------------------------------------------
// WMMA GEMM: out[rows x N] = act(A[rows x (K1+K2)] @ W + bias) (+resid)
// Wt is f16, transposed [N x K]. A split: cols [0,K1) from A1 (stride K1),
// cols [K1,K) from A2 (stride K2, row index row/grp if grp>0).
// One wave computes a 16x64 C tile (4 WMMA tiles sharing the A fragment).
// rows multiple of 16; K1, K multiples of 32.
// ---------------------------------------------------------------------------
__global__ __launch_bounds__(256)
void k_gemm(const float* __restrict__ A1, int K1,
            const float* __restrict__ A2, int K2, int grp,
            const _Float16* __restrict__ Wt, const float* __restrict__ bias,
            const float* __restrict__ resid,
            float* __restrict__ out, int rows, int N, int act)
{
  const int wave = threadIdx.x >> 5;
  const int lane = threadIdx.x & 31;
  const int r0 = blockIdx.x * 16;
  const int c0 = (blockIdx.y * 8 + wave) * 64;
  if (c0 >= N) return;                       // wave-uniform exit
  const int K  = K1 + K2;
  const int m  = lane & 15;                  // A row in tile / C column
  const int hh = lane >> 4;                  // half-wave select
  const int row = r0 + m;
  const int n0 = c0 + m;
  const int n1 = n0 + 16;
  const int n2 = n0 + 32;
  const int n3 = n0 + 48;
  auto clampn = [&](int n) { return (n < N) ? n : (N - 1); };
  const float* a1r = A1 + (size_t)row * K1;
  const float* a2r = A2 ? (A2 + (size_t)(grp ? row / grp : row) * K2) : a1r;
  const _Float16* w0 = Wt + (size_t)clampn(n0) * K + hh * 16;
  const _Float16* w1 = Wt + (size_t)clampn(n1) * K + hh * 16;
  const _Float16* w2 = Wt + (size_t)clampn(n2) * K + hh * 16;
  const _Float16* w3 = Wt + (size_t)clampn(n3) * K + hh * 16;

  v8f acc0 = {}, acc1 = {}, acc2 = {}, acc3 = {};

  // One K-step: A fragment layout (16-bit A 16x32): element e holds
  // K = hh*16+e (matches contiguous B) -> four contiguous float4 at
  // ap+hh*16.
  auto kstep = [&](const float* ap, int k0) {
    const float* p = ap + hh * 16;
    f32x4 x0 = *(const f32x4*)p;
    f32x4 x1 = *(const f32x4*)(p + 4);
    f32x4 x2 = *(const f32x4*)(p + 8);
    f32x4 x3 = *(const f32x4*)(p + 12);
    v16h av;
#pragma unroll
    for (int e = 0; e < 4; ++e) {
      av[e]      = (_Float16)x0[e];
      av[4 + e]  = (_Float16)x1[e];
      av[8 + e]  = (_Float16)x2[e];
      av[12 + e] = (_Float16)x3[e];
    }
    v16h b0 = *(const v16h*)(w0 + k0);
    v16h b1 = *(const v16h*)(w1 + k0);
    v16h b2 = *(const v16h*)(w2 + k0);
    v16h b3 = *(const v16h*)(w3 + k0);
    acc0 = __builtin_amdgcn_wmma_f32_16x16x32_f16(
        false, av, false, b0, (short)0, acc0, false, false);
    acc1 = __builtin_amdgcn_wmma_f32_16x16x32_f16(
        false, av, false, b1, (short)0, acc1, false, false);
    acc2 = __builtin_amdgcn_wmma_f32_16x16x32_f16(
        false, av, false, b2, (short)0, acc2, false, false);
    acc3 = __builtin_amdgcn_wmma_f32_16x16x32_f16(
        false, av, false, b3, (short)0, acc3, false, false);
  };

  // Split loops: no per-iteration A1/A2 selection in the hot path.
#pragma unroll 2
  for (int k0 = 0; k0 < K1; k0 += 32) kstep(a1r + k0, k0);
#pragma unroll 2
  for (int k0 = K1; k0 < K; k0 += 32) kstep(a2r + (k0 - K1), k0);

  auto epi = [&](const v8f& a, int n) {
    if (n >= N) return;
    const float bb = bias ? bias[n] : 0.0f;
#pragma unroll
    for (int v = 0; v < 8; ++v) {
      int rr = r0 + v + 8 * hh;              // C/D: VGPR v -> row v (+8)
      float val = a[v] + bb;
      if (act == 1) val = geluf(val);
      size_t o = (size_t)rr * N + n;
      if (resid) val += resid[o];
      out[o] = val;
    }
  };
  epi(acc0, n0);
  epi(acc1, n1);
  epi(acc2, n2);
  epi(acc3, n3);
}

// GEMV for N==1 projections (gates / scores): one wave per row.
__global__ __launch_bounds__(256)
void k_gemv(const float* __restrict__ A1, int K1,
            const float* __restrict__ A2, int K2, int grp,
            const float* __restrict__ w, const float* __restrict__ b,
            float* __restrict__ out, int rows, int act)
{
  int row  = blockIdx.x * 8 + (threadIdx.x >> 5);
  int lane = threadIdx.x & 31;
  if (row >= rows) return;
  int K = K1 + K2;
  const float* a1 = A1 + (size_t)row * K1;
  const float* a2 = A2 ? (A2 + (size_t)(grp ? row / grp : row) * K2) : a1;
  float s = 0.f;
  for (int k = lane; k < K; k += 32)
    s += ((k < K1) ? a1[k] : a2[k - K1]) * w[k];
  for (int off = 16; off; off >>= 1) s += __shfl_xor(s, off, 32);
  if (lane == 0) {
    float v = s + (b ? b[0] : 0.f);
    if (act == 1) v = geluf(v);
    else if (act == 2) v = halfsigf(v);
    out[row] = v;
  }
}

// LayerNorm over last dim; one block per row; safe in-place.
__global__ __launch_bounds__(256)
void k_ln(const float* __restrict__ x, const float* __restrict__ g,
          const float* __restrict__ b, float* __restrict__ out, int D)
{
  int row = blockIdx.x, tid = threadIdx.x;
  const float* xr = x + (size_t)row * D;
  float s = 0.f, s2 = 0.f;
  for (int i = tid; i < D; i += 256) { float v = xr[i]; s += v; s2 += v * v; }
  __shared__ float sh[256], sh2[256];
  sh[tid] = s; sh2[tid] = s2; __syncthreads();
  for (int st = 128; st; st >>= 1) {
    if (tid < st) { sh[tid] += sh[tid + st]; sh2[tid] += sh2[tid + st]; }
    __syncthreads();
  }
  float mean = sh[0] / D;
  float var  = sh2[0] / D - mean * mean;
  float rst  = rsqrtf(var + 1e-5f);
  for (int i = tid; i < D; i += 256)
    out[(size_t)row * D + i] = (xr[i] - mean) * rst * g[i] + b[i];
}

// Depthwise 1D conv ('NWC','WIO', groups=D) + bias (+ optional exact GELU).
__global__ __launch_bounds__(256)
void k_dwconv(const float* __restrict__ x, const float* __restrict__ k,
              const float* __restrict__ b, float* __restrict__ y,
              int Bb, int L, int Lo, int D, int KW, int stride, int pad,
              int dogelu)
{
  size_t i = (size_t)blockIdx.x * blockDim.x + threadIdx.x;
  size_t tot = (size_t)Bb * Lo * D;
  if (i >= tot) return;
  int d  = (int)(i % D);
  int lo = (int)((i / D) % Lo);
  int bb = (int)(i / ((size_t)D * Lo));
  float s = 0.f;
  for (int j = 0; j < KW; ++j) {
    int li = lo * stride - pad + j;
    if (li >= 0 && li < L) s += x[((size_t)bb * L + li) * D + d] * k[j * D + d];
  }
  s += b[d];
  if (dogelu) s = geluf(s);
  y[i] = s;
}

// summ = concat(mean over L, max over L) -> (B, 2D)
__global__ __launch_bounds__(256)
void k_poolmm(const float* __restrict__ x, float* __restrict__ out,
              int Bb, int L, int D)
{
  int i = blockIdx.x * blockDim.x + threadIdx.x;
  if (i >= Bb * D) return;
  int d = i % D, bb = i / D;
  const float* xp = x + (size_t)bb * L * D + d;
  float s = 0.f, mx = -3.4e38f;
  for (int t = 0; t < L; ++t) { float v = xp[(size_t)t * D]; s += v; mx = fmaxf(mx, v); }
  out[(size_t)bb * 2 * D + d]     = s / (float)L;
  out[(size_t)bb * 2 * D + D + d] = mx;
}

// pooled[b,i,:] = mean of x[b,2i,:], x[b,2i+1,:]
__global__ __launch_bounds__(256)
void k_pairs(const float* __restrict__ x, float* __restrict__ pooled,
             int Bb, int L, int D)
{
  int m = L / 2;
  size_t i = (size_t)blockIdx.x * blockDim.x + threadIdx.x;
  size_t tot = (size_t)Bb * m * D;
  if (i >= tot) return;
  int d = (int)(i % D);
  int t = (int)((i / D) % m);
  int bb = (int)(i / ((size_t)D * m));
  pooled[i] = 0.5f * (x[((size_t)bb * L + 2 * t) * D + d] +
                      x[((size_t)bb * L + 2 * t + 1) * D + d]);
}

// merged rows [8, 8+m): pooled + gate*(r - pooled)
__global__ __launch_bounds__(256)
void k_merge(const float* __restrict__ pooled, const float* __restrict__ r,
             const float* __restrict__ gate, float* __restrict__ outlv,
             int Bb, int m, int D)
{
  size_t i = (size_t)blockIdx.x * blockDim.x + threadIdx.x;
  size_t tot = (size_t)Bb * m * D;
  if (i >= tot) return;
  int d = (int)(i % D);
  int t = (int)((i / D) % m);
  int bb = (int)(i / ((size_t)D * m));
  size_t ri = ((size_t)bb * m + t) * D + d;
  float p = pooled[ri], rv = r[ri];
  outlv[((size_t)bb * (m + 8) + 8 + t) * D + d] = p + gate[(size_t)bb * m + t] * (rv - p);
}

// mem += wg * (cand - mem), wg per (b,j)
__global__ __launch_bounds__(256)
void k_memupd(float* __restrict__ mem, const float* __restrict__ cand,
              const float* __restrict__ wg, int rows, int D)
{
  size_t i = (size_t)blockIdx.x * blockDim.x + threadIdx.x;
  if (i >= (size_t)rows * D) return;
  size_t r = i / D;
  mem[i] += wg[r] * (cand[i] - mem[i]);
}

// rs[b,:] = mean over M of mem[b,:,:]
__global__ __launch_bounds__(256)
void k_groupmean(const float* __restrict__ mem, float* __restrict__ rs,
                 int Bb, int M_, int D)
{
  int i = blockIdx.x * blockDim.x + threadIdx.x;
  if (i >= Bb * D) return;
  int d = i % D, bb = i / D;
  float s = 0.f;
  for (int j = 0; j < M_; ++j) s += mem[((size_t)bb * M_ + j) * D + d];
  rs[(size_t)bb * D + d] = s / (float)M_;
}

__global__ __launch_bounds__(256)
void k_bcast(const float* __restrict__ src, float* __restrict__ dst,
             int per, int tot)
{
  int i = blockIdx.x * blockDim.x + threadIdx.x;
  if (i < tot) dst[i] = src[i % per];
}

// Per-batch top-8 of scores (first-occurrence ties), sorted ascending;
// gather carrier rows of x into merged rows [0,8).
__global__ __launch_bounds__(256)
void k_topk(const float* __restrict__ scores, const float* __restrict__ x,
            float* __restrict__ outlv, int L, int D, int m)
{
  int bb = blockIdx.x, tid = threadIdx.x;
  __shared__ float sc[1024];
  __shared__ float rv[256];
  __shared__ int   ri[256];
  __shared__ int   sel[8];
  for (int i = tid; i < L; i += 256) sc[i] = scores[(size_t)bb * L + i];
  __syncthreads();
  for (int t = 0; t < 8; ++t) {
    float bv = -3.4e38f; int bi = 0x7fffffff;
    for (int i = tid; i < L; i += 256) {
      float v = sc[i];
      if (v > bv || (v == bv && i < bi)) { bv = v; bi = i; }
    }
    rv[tid] = bv; ri[tid] = bi;
    __syncthreads();
    for (int s = 128; s; s >>= 1) {
      if (tid < s) {
        if (rv[tid + s] > rv[tid] ||
            (rv[tid + s] == rv[tid] && ri[tid + s] < ri[tid])) {
          rv[tid] = rv[tid + s]; ri[tid] = ri[tid + s];
        }
      }
      __syncthreads();
    }
    if (tid == 0) { sel[t] = ri[0]; sc[ri[0]] = -3.4e38f; }
    __syncthreads();
  }
  if (tid == 0) {                    // sort 8 indices ascending
    for (int a = 1; a < 8; ++a) {
      int v = sel[a], c = a;
      while (c > 0 && sel[c - 1] > v) { sel[c] = sel[c - 1]; --c; }
      sel[c] = v;
    }
  }
  __syncthreads();
  for (int j = 0; j < 8; ++j)
    for (int d = tid; d < D; d += 256)
      outlv[((size_t)bb * (m + 8) + j) * D + d] =
          x[((size_t)bb * L + sel[j]) * D + d];
}

// readout: pooled[:, li*D:(li+1)*D] = hidden * g ; level_gates[:, li] = g
__global__ __launch_bounds__(256)
void k_ro_scale(const float* __restrict__ hidden, const float* __restrict__ g,
                float* __restrict__ pooled, float* __restrict__ gates_out,
                int li, int D)
{
  int i = blockIdx.x * blockDim.x + threadIdx.x;
  if (i >= 32 * D) return;
  int d = i % D, bb = i / D;
  pooled[(size_t)bb * (5 * D) + li * D + d] = hidden[(size_t)bb * D + d] * g[bb];
  if (d == 0) gates_out[bb * 5 + li] = g[bb];
}

// weights[b, off : off+Ll] = g[b] / Ll
__global__ __launch_bounds__(256)
void k_ro_weights(const float* __restrict__ g, float* __restrict__ wout,
                  int off, int Ll, int totW)
{
  int i = blockIdx.x * blockDim.x + threadIdx.x;
  if (i >= 32 * Ll) return;
  int t = i % Ll, bb = i / Ll;
  wout[(size_t)bb * totW + off + t] = g[bb] / (float)Ll;
}

// ---------------------------------------------------------------------------
// Host orchestration
// ---------------------------------------------------------------------------
enum {
  IN_LAT = 0,
  R_SN_G, R_SN_B, R_DW_K, R_DW_B, R_PW_W, R_PW_B, R_CN_G, R_CN_B,
  R_M_W1, R_M_B1, R_M_W2, R_M_B2,
  S_SN_G, S_SN_B, S_DW_K, S_DW_B, S_PW_W, S_PW_B, S_MG_W, S_MG_B,
  S_CS_W, S_CS_B, S_ON_G, S_ON_B,
  G_MEM0, G_SP_W, G_SP_B, G_WG_W, G_WG_B, G_WC_W1, G_WC_B1, G_WC_W2, G_WC_B2,
  G_RP_W1, G_RP_B1, G_RP_W2, G_RP_B2, G_MN_G, G_MN_B, G_LN_G, G_LN_B,
  O_LS_W, O_LS_B, O_LG_W, O_LG_B, O_ON_G, O_ON_B, O_CLS_W, O_CLS_B
};

extern "C" void kernel_launch(void* const* d_in, const int* in_sizes, int n_in,
                              void* d_out, int out_size, void* d_ws, size_t ws_size,
                              hipStream_t stream)
{
  (void)in_sizes; (void)n_in; (void)out_size; (void)ws_size;
  constexpr int Bn = 32, Dd = 512, Hh = 2048, Mm = 32;
  constexpr int CHUNK = 8192;

  auto F = [&](int i) { return (const float*)d_in[i]; };

  // ---- workspace bump allocator (fp32 region; all sizes 32-float aligned) ----
  float* ws = (float*)d_ws;
  size_t off = 0;
  auto alloc = [&](size_t n) { float* p = ws + off; off += n; return p; };
  const size_t RMAX = (size_t)Bn * 1024;
  float* xbuf   = alloc(RMAX * Dd);                 // working x
  float* nbuf   = alloc(RMAX * Dd);                 // LN output
  float* hbuf   = alloc(RMAX * Dd);                 // dwconv / hidden
  float* c1buf  = alloc((size_t)CHUNK * Hh);        // MLP H-chunk
  float* lvbuf  = alloc((size_t)Bn * 1009 * Dd);    // levels 1..4 storage
  float* rbuf   = alloc((size_t)Bn * 512 * Dd);     // reduce r
  float* pbuf   = alloc((size_t)Bn * 512 * Dd);     // reduce pooled pairs
  float* membuf = alloc((size_t)Bn * Mm * Dd);
  float* nmbuf  = alloc((size_t)Bn * Mm * Dd);
  float* chbuf  = alloc((size_t)Bn * Mm * Dd);      // gelu(wi@w1)
  float* cdbuf  = alloc((size_t)Bn * Mm * Dd);      // cand
  float* summb  = alloc((size_t)Bn * 2 * Dd);
  float* spb    = alloc((size_t)Bn * Dd);           // gelu(summ@g_sp_w)
  float* rsb    = alloc((size_t)Bn * Dd);           // mem mean
  float* wgb    = alloc((size_t)Bn * Mm);
  float* scb    = alloc((size_t)Bn * 1024);         // scores
  float* gtb    = alloc((size_t)Bn * 512);          // reduce gate
  float* hidb   = alloc((size_t)Bn * Dd);
  float* gb     = alloc((size_t)Bn);
  float* poolb  = alloc((size_t)Bn * 5 * Dd);       // readout pooled (32x2560)

  // ---- f16 transposed-weight region (32B aligned: off is 32-float mult) ----
  _Float16* wsh = (_Float16*)(ws + off);
  size_t hoff = 0;
  auto halloc = [&](size_t n) { _Float16* p = wsh + hoff; hoff += n; return p; };
  _Float16* t_pw  = halloc((size_t)512 * 512);
  _Float16* t_m1  = halloc((size_t)512 * 2048);
  _Float16* t_m2  = halloc((size_t)2048 * 512);
  _Float16* t_spw = halloc((size_t)512 * 512);
  _Float16* t_sp  = halloc((size_t)1024 * 512);
  _Float16* t_wc1 = halloc((size_t)1024 * 512);
  _Float16* t_wc2 = halloc((size_t)512 * 512);
  _Float16* t_rp1 = halloc((size_t)1024 * 512);
  _Float16* t_rp2 = halloc((size_t)512 * 512);
  _Float16* t_ls  = halloc((size_t)1024 * 512);
  _Float16* t_cls = halloc((size_t)2560 * 1000);

  // ---- output layout (tuple flattened in return order) ----
  float* out        = (float*)d_out;
  float* out_logits = out;                          // 32*1000
  float* out_w      = out + 32000;                  // 32*2033
  float* out_lg     = out + 97056;                  // 32*5
  float* out_rg[4]  = { out + 97216, out + 113600, out + 121920, out + 126208 };

  // ---- launch helpers ----
  auto ew = [&](size_t total) { return dim3((unsigned)((total + 255) / 256)); };
  auto wcvt = [&](int wi, _Float16* t, int K, int N) {
    k_wcvt<<<ew((size_t)K * N), 256, 0, stream>>>(F(wi), t, K, N);
  };
  auto ln = [&](const float* x, int gi, int bi, float* o, int rows, int D) {
    k_ln<<<rows, 256, 0, stream>>>(x, F(gi), F(bi), o, D);
  };
  auto gemm = [&](const float* A1, int K1, const float* A2, int K2, int grp,
                  const _Float16* Wt, int Bi, const float* resid, float* o,
                  int rows, int N, int act) {
    dim3 g((rows + 15) / 16, (((N + 63) / 64) + 7) / 8);
    k_gemm<<<g, 256, 0, stream>>>(A1, K1, A2, K2, grp, Wt, F(Bi),
                                  resid, o, rows, N, act);
  };
  auto gemv = [&](const float* A1, int K1, const float* A2, int K2, int grp,
                  int wi, int bi, float* o, int rows, int act) {
    k_gemv<<<(rows + 7) / 8, 256, 0, stream>>>(A1, K1, A2, K2, grp,
                                               F(wi), F(bi), o, rows, act);
  };

  // ---- weight transpose/convert (once per launch, ~16MB, L2-resident) ----
  wcvt(R_PW_W,  t_pw,  512, 512);
  wcvt(R_M_W1,  t_m1,  512, 2048);
  wcvt(R_M_W2,  t_m2,  2048, 512);
  wcvt(S_PW_W,  t_spw, 512, 512);
  wcvt(G_SP_W,  t_sp,  1024, 512);
  wcvt(G_WC_W1, t_wc1, 1024, 512);
  wcvt(G_WC_W2, t_wc2, 512, 512);
  wcvt(G_RP_W1, t_rp1, 1024, 512);
  wcvt(G_RP_W2, t_rp2, 512, 512);
  wcvt(O_LS_W,  t_ls,  1024, 512);
  wcvt(O_CLS_W, t_cls, 2560, 1000);

  auto refine = [&](float* x, int L) {
    int rows = Bn * L;
    ln(x, R_SN_G, R_SN_B, nbuf, rows, Dd);
    k_dwconv<<<ew((size_t)rows * Dd), 256, 0, stream>>>(
        nbuf, F(R_DW_K), F(R_DW_B), hbuf, Bn, L, L, Dd, 5, 1, 2, 1);
    gemm(hbuf, Dd, nullptr, 0, 0, t_pw, R_PW_B, x, x, rows, Dd, 0);
    ln(x, R_CN_G, R_CN_B, nbuf, rows, Dd);
    for (int r0 = 0; r0 < rows; r0 += CHUNK) {
      int cr = rows - r0 < CHUNK ? rows - r0 : CHUNK;
      gemm(nbuf + (size_t)r0 * Dd, Dd, nullptr, 0, 0, t_m1, R_M_B1,
           nullptr, c1buf, cr, Hh, 1);
      gemm(c1buf, Hh, nullptr, 0, 0, t_m2, R_M_B2,
           x + (size_t)r0 * Dd, x + (size_t)r0 * Dd, cr, Dd, 0);
    }
  };

  auto memstep = [&](float* x, int L) {
    int rows = Bn * L;
    k_poolmm<<<ew(Bn * Dd), 256, 0, stream>>>(x, summb, Bn, L, Dd);
    gemm(summb, 2 * Dd, nullptr, 0, 0, t_sp, G_SP_B, nullptr, spb, Bn, Dd, 1);
    ln(membuf, G_MN_G, G_MN_B, nmbuf, Bn * Mm, Dd);
    gemv(nmbuf, Dd, spb, Dd, Mm, G_WG_W, G_WG_B, wgb, Bn * Mm, 2);
    gemm(nmbuf, Dd, spb, Dd, Mm, t_wc1, G_WC_B1, nullptr, chbuf, Bn * Mm, Dd, 1);
    gemm(chbuf, Dd, nullptr, 0, 0, t_wc2, G_WC_B2, nullptr, cdbuf, Bn * Mm, Dd, 0);
    k_memupd<<<ew((size_t)Bn * Mm * Dd), 256, 0, stream>>>(membuf, cdbuf, wgb,
                                                           Bn * Mm, Dd);
    k_groupmean<<<ew(Bn * Dd), 256, 0, stream>>>(membuf, rsb, Bn, Mm, Dd);
    ln(x, G_LN_G, G_LN_B, nbuf, rows, Dd);
    gemm(nbuf, Dd, rsb, Dd, L, t_rp1, G_RP_B1, nullptr, hbuf, rows, Dd, 1);
    gemm(hbuf, Dd, nullptr, 0, 0, t_rp2, G_RP_B2, x, x, rows, Dd, 0);
  };

  // ---- init: x = latents, mem = broadcast(g_mem0) ----
  hipMemcpyAsync(xbuf, d_in[IN_LAT], (size_t)Bn * 1024 * Dd * sizeof(float),
                 hipMemcpyDeviceToDevice, stream);
  k_bcast<<<ew((size_t)Bn * Mm * Dd), 256, 0, stream>>>(F(G_MEM0), membuf,
                                                        Mm * Dd, Bn * Mm * Dd);

  int Ls[5] = {1024, 0, 0, 0, 0};
  const float* lvl_ptr[5];
  lvl_ptr[0] = F(IN_LAT);
  size_t lvoff = 0;
  int L = 1024;

  for (int t = 0; t < 4; ++t) {
    refine(xbuf, L);
    memstep(xbuf, L);
    // ---- reduce ----
    int rows = Bn * L, m = L / 2, newL = m + 8;
    float* lv = lvbuf + lvoff;
    ln(xbuf, S_SN_G, S_SN_B, nbuf, rows, Dd);
    k_dwconv<<<ew((size_t)Bn * m * Dd), 256, 0, stream>>>(
        nbuf, F(S_DW_K), F(S_DW_B), hbuf, Bn, L, m, Dd, 3, 2, 1, 1);
    gemm(hbuf, Dd, nullptr, 0, 0, t_spw, S_PW_B, nullptr, rbuf, Bn * m, Dd, 0);
    k_pairs<<<ew((size_t)Bn * m * Dd), 256, 0, stream>>>(xbuf, pbuf, Bn, L, Dd);
    gemv(rbuf, Dd, pbuf, Dd, 0, S_MG_W, S_MG_B, gtb, Bn * m, 2);
    hipMemcpyAsync(out_rg[t], gtb, (size_t)Bn * m * sizeof(float),
                   hipMemcpyDeviceToDevice, stream);
    k_merge<<<ew((size_t)Bn * m * Dd), 256, 0, stream>>>(pbuf, rbuf, gtb, lv,
                                                         Bn, m, Dd);
    gemv(nbuf, Dd, nullptr, 0, 0, S_CS_W, S_CS_B, scb, rows, 0);
    k_topk<<<Bn, 256, 0, stream>>>(scb, xbuf, lv, L, Dd, m);
    ln(lv, S_ON_G, S_ON_B, lv, Bn * newL, Dd);
    lvl_ptr[t + 1] = lv;
    Ls[t + 1] = newL;
    hipMemcpyAsync(xbuf, lv, (size_t)Bn * newL * Dd * sizeof(float),
                   hipMemcpyDeviceToDevice, stream);
    lvoff += (size_t)Bn * newL * Dd;
    L = newL;
  }

  // final refine + memory; levels[-1] = final x
  refine(xbuf, L);
  memstep(xbuf, L);
  hipMemcpyAsync((void*)lvl_ptr[4], xbuf, (size_t)Bn * L * Dd * sizeof(float),
                 hipMemcpyDeviceToDevice, stream);

  // ---- readout ----
  const int woff[5] = {0, 1024, 1544, 1812, 1954};
  for (int li = 0; li < 5; ++li) {
    k_poolmm<<<ew(Bn * Dd), 256, 0, stream>>>(lvl_ptr[li], summb, Bn, Ls[li], Dd);
    gemm(summb, 2 * Dd, nullptr, 0, 0, t_ls, O_LS_B, nullptr, hidb, Bn, Dd, 1);
    gemv(hidb, Dd, nullptr, 0, 0, O_LG_W, O_LG_B, gb, Bn, 2);
    k_ro_scale<<<ew(Bn * Dd), 256, 0, stream>>>(hidb, gb, poolb, out_lg, li, Dd);
    k_ro_weights<<<ew((size_t)Bn * Ls[li]), 256, 0, stream>>>(gb, out_w,
                                                              woff[li], Ls[li], 2033);
  }
  ln(poolb, O_ON_G, O_ON_B, poolb, Bn, 5 * Dd);
  gemm(poolb, 5 * Dd, nullptr, 0, 0, t_cls, O_CLS_B, nullptr,
       out_logits, Bn, 1000, 0);
}
